// BertLSHSelfAttention_88665304859337
// MI455X (gfx1250) — matrix operand order repaired and checked by reading, not compiled, not run on hardware
//
#include <hip/hip_runtime.h>
#include <hip/hip_bf16.h>
#include <cstdint>

// ---------------------------------------------------------------------------
// BERT LSH self-attention for gfx1250 (MI455X), wave32 + WMMA bf16 path.
// B=2, S=1024, D=768, H=12, dh=64, NH=4, BANDS=8, TABLE=64
// Software-pipelined WMMA loops (loads for step k+1 issued before step k's
// matrix ops) so v_wmma overlaps outstanding global_load_b128.
// ---------------------------------------------------------------------------

#define B_  2
#define S_  1024
#define D_  768
#define H_  12
#define DH_ 64
#define NH_ 4
#define BANDS_ 8

typedef __attribute__((ext_vector_type(16))) __bf16 v16bf;
typedef __attribute__((ext_vector_type(8)))  float  v8f;

__device__ __forceinline__ unsigned short f2bf(float f) {
    uint32_t u = __float_as_uint(f);
    u += 0x7FFFu + ((u >> 16) & 1u);          // round-to-nearest-even
    return (unsigned short)(u >> 16);
}
__device__ __forceinline__ float bf2f(unsigned short u) {
    return __uint_as_float(((uint32_t)u) << 16);
}

union Frag16 { uint4 q[2]; v16bf v; };

// A-operand (16x32 bf16, M=lane%16, K-major rows):
// lanes 0-15 hold K = kb..kb+7 and kb+16..kb+23 ; lanes 16-31 shifted by 8.
__device__ __forceinline__ v16bf load_frag_a(const unsigned short* row, int kbase, int lane) {
    int kb = kbase + ((lane & 16) ? 8 : 0);
    Frag16 f;
    f.q[0] = *(const uint4*)(row + kb);
    f.q[1] = *(const uint4*)(row + kb + 16);
    return f.v;
}
// B-operand (32x16 bf16, N=lane%16, K-major rows):
// lanes 0-15 hold K = kb..kb+7 and kb+16..kb+23 ; lanes 16-31 shifted by 8.
__device__ __forceinline__ v16bf load_frag_b(const unsigned short* row, int kbase, int lane) {
    int kb = kbase + ((lane & 16) ? 8 : 0);
    Frag16 f;
    f.q[0] = *(const uint4*)(row + kb);
    f.q[1] = *(const uint4*)(row + kb + 16);
    return f.v;
}

#define WMMA_BF16(A, B, C) \
    __builtin_amdgcn_wmma_f32_16x16x32_bf16(false, (A), false, (B), (short)0, (C), false, false)

// ---------------------------------------------------------------------------
// Kernel 1: f32 -> bf16 staging, vectorized (n must be a multiple of 4).
// ---------------------------------------------------------------------------
__global__ void convert_bf16(const float4* __restrict__ src,
                             ushort4* __restrict__ dst, int n4) {
    int i = blockIdx.x * blockDim.x + threadIdx.x;
    if (i < n4) {
        float4 v = src[i];
        ushort4 o;
        o.x = f2bf(v.x); o.y = f2bf(v.y); o.z = f2bf(v.z); o.w = f2bf(v.w);
        dst[i] = o;
    }
}

// ---------------------------------------------------------------------------
// Kernel 2: fused QKV projection GEMM.  out = X @ W^T + b  (per z in {q,k,v})
// Grid: (D_/64, (B_*S_)/128, 3). Block: 128 (4 waves); wave -> 32x64 tile,
// 8 accumulators, 8 WMMAs per 32-wide k-step, k+32 fragments prefetched.
// ---------------------------------------------------------------------------
__global__ __launch_bounds__(128) void qkv_gemm(
    const unsigned short* __restrict__ xb,   // [B*S, D] bf16
    const unsigned short* __restrict__ wqb,  // [D, D] bf16 (row n = w_q[n,:])
    const unsigned short* __restrict__ wkb,
    const unsigned short* __restrict__ wvb,
    const float* __restrict__ bq, const float* __restrict__ bk,
    const float* __restrict__ bv,
    float* __restrict__ qf, float* __restrict__ kf,
    unsigned short* __restrict__ qb16, unsigned short* __restrict__ kb16,
    unsigned short* __restrict__ vt)
{
    const int lane = threadIdx.x & 31;
    const int wave = threadIdx.x >> 5;
    const int z = blockIdx.z;
    const int n0 = blockIdx.x * 64;
    const int m0 = blockIdx.y * 128 + wave * 32;

    const unsigned short* W   = (z == 0) ? wqb : (z == 1) ? wkb : wvb;
    const float*          bia = (z == 0) ? bq  : (z == 1) ? bk  : bv;

    const unsigned short* arow0 = xb + (size_t)(m0 + (lane & 15)) * D_;
    const unsigned short* arow1 = arow0 + (size_t)16 * D_;
    const unsigned short* brow  = W  + (size_t)(n0 + (lane & 15)) * D_;

    v8f acc[2][4];
    #pragma unroll
    for (int mi = 0; mi < 2; mi++)
        #pragma unroll
        for (int ci = 0; ci < 4; ci++) acc[mi][ci] = v8f{};

    // prologue: fragments for k = 0
    v16bf a0 = load_frag_a(arow0, 0, lane);
    v16bf a1 = load_frag_a(arow1, 0, lane);
    v16bf b0 = load_frag_b(brow,             0, lane);
    v16bf b1 = load_frag_b(brow + 16 * D_,   0, lane);
    v16bf b2 = load_frag_b(brow + 32 * D_,   0, lane);
    v16bf b3 = load_frag_b(brow + 48 * D_,   0, lane);

    // main loop: prefetch k+32 fragments, then 8 WMMAs on current fragments
    for (int k = 0; k < D_ - 32; k += 32) {
        const int kn = k + 32;
        v16bf na0 = load_frag_a(arow0, kn, lane);
        v16bf na1 = load_frag_a(arow1, kn, lane);
        v16bf nb0 = load_frag_b(brow,            kn, lane);
        v16bf nb1 = load_frag_b(brow + 16 * D_,  kn, lane);
        v16bf nb2 = load_frag_b(brow + 32 * D_,  kn, lane);
        v16bf nb3 = load_frag_b(brow + 48 * D_,  kn, lane);

        acc[0][0] = WMMA_BF16(a0, b0, acc[0][0]);
        acc[0][1] = WMMA_BF16(a0, b1, acc[0][1]);
        acc[0][2] = WMMA_BF16(a0, b2, acc[0][2]);
        acc[0][3] = WMMA_BF16(a0, b3, acc[0][3]);
        acc[1][0] = WMMA_BF16(a1, b0, acc[1][0]);
        acc[1][1] = WMMA_BF16(a1, b1, acc[1][1]);
        acc[1][2] = WMMA_BF16(a1, b2, acc[1][2]);
        acc[1][3] = WMMA_BF16(a1, b3, acc[1][3]);

        a0 = na0; a1 = na1; b0 = nb0; b1 = nb1; b2 = nb2; b3 = nb3;
    }
    // epilogue: last k-step
    acc[0][0] = WMMA_BF16(a0, b0, acc[0][0]);
    acc[0][1] = WMMA_BF16(a0, b1, acc[0][1]);
    acc[0][2] = WMMA_BF16(a0, b2, acc[0][2]);
    acc[0][3] = WMMA_BF16(a0, b3, acc[0][3]);
    acc[1][0] = WMMA_BF16(a1, b0, acc[1][0]);
    acc[1][1] = WMMA_BF16(a1, b1, acc[1][1]);
    acc[1][2] = WMMA_BF16(a1, b2, acc[1][2]);
    acc[1][3] = WMMA_BF16(a1, b3, acc[1][3]);

    const int iBase = (lane & 16) ? 8 : 0;
    #pragma unroll
    for (int mi = 0; mi < 2; mi++) {
        #pragma unroll
        for (int ci = 0; ci < 4; ci++) {
            const int n = n0 + ci * 16 + (lane & 15);  // output column 0..767
            const int h = n >> 6, d = n & 63;
            const float bval = bia[n];
            #pragma unroll
            for (int r = 0; r < 8; r++) {
                const int m = m0 + mi * 16 + iBase + r;  // output row 0..2047
                const int b = m >> 10, s = m & (S_ - 1);
                const float v = acc[mi][ci][r] + bval;
                const size_t idx = (((size_t)b * H_ + h) * S_ + s) * DH_ + d;
                if (z == 0)      { qf[idx] = v; qb16[idx] = f2bf(v); }
                else if (z == 1) { kf[idx] = v; kb16[idx] = f2bf(v); }
                else { vt[(((size_t)b * H_ + h) * DH_ + d) * S_ + s] = f2bf(v); }
            }
        }
    }
}

// ---------------------------------------------------------------------------
// Kernel 3: LSH hashing. One thread per (b,h,s). A 128-thread block covers
// 128 consecutive s of one (b,h): stage the shared random vectors (8 KB) and
// coefficients in LDS once per block.
// ---------------------------------------------------------------------------
__global__ __launch_bounds__(128) void lsh_hash(
    const float* __restrict__ qf, const float* __restrict__ kf,
    const float* __restrict__ rv,   // [B,H,NH,BANDS,dh]
    const int* __restrict__ hc,     // [B,H,NH,BANDS]
    uint32_t* __restrict__ hqp, uint32_t* __restrict__ hkp)
{
    __shared__ float s_rv[NH_ * BANDS_ * DH_];   // 8 KB
    __shared__ int   s_hc[NH_ * BANDS_];

    const int t = blockIdx.x * 128 + threadIdx.x;
    const int bh = t >> 10;                       // b*H + h (S = 1024)

    // cooperative staging (same bh for whole block)
    const float4* rvb4 = (const float4*)(rv + (size_t)bh * NH_ * BANDS_ * DH_);
    float4* s_rv4 = (float4*)s_rv;
    #pragma unroll
    for (int i = threadIdx.x; i < NH_ * BANDS_ * DH_ / 4; i += 128)
        s_rv4[i] = rvb4[i];
    if (threadIdx.x < NH_ * BANDS_)
        s_hc[threadIdx.x] = hc[(size_t)bh * NH_ * BANDS_ + threadIdx.x];
    __syncthreads();

    const float4* q4 = (const float4*)(qf + (size_t)t * DH_);
    const float4* k4 = (const float4*)(kf + (size_t)t * DH_);
    float qr[DH_], kr[DH_];
    #pragma unroll
    for (int d4 = 0; d4 < DH_ / 4; d4++) {
        float4 qv = q4[d4], kv = k4[d4];
        qr[d4 * 4 + 0] = qv.x; qr[d4 * 4 + 1] = qv.y;
        qr[d4 * 4 + 2] = qv.z; qr[d4 * 4 + 3] = qv.w;
        kr[d4 * 4 + 0] = kv.x; kr[d4 * 4 + 1] = kv.y;
        kr[d4 * 4 + 2] = kv.z; kr[d4 * 4 + 3] = kv.w;
    }

    uint32_t hq = 0, hk = 0;
    for (int n = 0; n < NH_; n++) {
        int aq = 0, ak = 0;
        for (int band = 0; band < BANDS_; band++) {
            const float* r = &s_rv[(n * BANDS_ + band) * DH_];
            float dq = 0.f, dk = 0.f;
            #pragma unroll
            for (int d = 0; d < DH_; d++) {
                dq = fmaf(qr[d], r[d], dq);
                dk = fmaf(kr[d], r[d], dk);
            }
            const int c = s_hc[n * BANDS_ + band];
            if (dq > 0.f) aq += c;
            if (dk > 0.f) ak += c;
        }
        hq |= (uint32_t)(aq & 63) << (8 * n);
        hk |= (uint32_t)(ak & 63) << (8 * n);
    }
    hqp[t] = hq; hkp[t] = hk;
}

// any-byte-equal: true iff some byte of a equals the same-position byte of b
__device__ __forceinline__ bool any_byte_eq(uint32_t a, uint32_t b) {
    uint32_t x = a ^ b;
    return ((x - 0x01010101u) & ~x & 0x80808080u) != 0u;
}

// ---------------------------------------------------------------------------
// Kernel 4: masked attention. Grid (S/16, H, B); block 128 (4 waves).
// Phase 1: scores via WMMA, K-tile fragments double-buffered; collision mask
//          (triu-symmetrized, A=0 where no collision), *1/8, bf16 -> LDS.
// Phase 2: full-row softmax (zeros participate, per reference) via shfl.
// Phase 3: P@V via WMMA; probs (A) from LDS, V^T (B) double-buffered global.
// ---------------------------------------------------------------------------
__global__ __launch_bounds__(128) void lsh_attention(
    const unsigned short* __restrict__ qb16,  // [B,H,S,dh] bf16
    const unsigned short* __restrict__ kb16,  // [B,H,S,dh] bf16
    const unsigned short* __restrict__ vt,    // [B,H,dh,S] bf16
    const uint32_t* __restrict__ hqp, const uint32_t* __restrict__ hkp,
    float* __restrict__ out)                  // [B,S,D] f32
{
    __shared__ __align__(16) unsigned short sc16[16][S_ + 16];  // 33 KB LDS

    const int lane = threadIdx.x & 31;
    const int wave = threadIdx.x >> 5;
    const int i0 = blockIdx.x * 16;
    const int h  = blockIdx.y;
    const int b  = blockIdx.z;
    const size_t bh = (size_t)b * H_ + h;
    const uint32_t* hq = hqp + bh * S_;
    const uint32_t* hk = hkp + bh * S_;
    const int iBase = (lane & 16) ? 8 : 0;

    // wave's V^T row for phase 3; prefetch it now so it overlaps phases 1-2
    const int d0 = wave * 16;
    const unsigned short* vrow = vt + (bh * DH_ + d0 + (lane & 15)) * S_;
    #pragma unroll
    for (int p = 0; p < 8; p++)
        __builtin_prefetch(vrow + p * 128, 0, 1);   // global_prefetch_b8

    // Q fragments for this 16-query tile (dh = 64 -> two 16x32 fragments)
    const unsigned short* qrow = qb16 + (bh * S_ + i0 + (lane & 15)) * DH_;
    const v16bf qa0 = load_frag_a(qrow, 0,  lane);
    const v16bf qa1 = load_frag_a(qrow, 32, lane);

    // loop-invariant query-row hashes
    uint32_t hqi[8], hki[8];
    #pragma unroll
    for (int r = 0; r < 8; r++) {
        const int i = i0 + iBase + r;
        hqi[r] = hq[i]; hki[r] = hk[i];
    }

    // ---- Phase 1: masked scores into LDS (K fragments double-buffered) ----
    {
        int jt = wave;
        const unsigned short* krow =
            kb16 + (bh * S_ + jt * 16 + (lane & 15)) * DH_;
        v16bf kb0 = load_frag_b(krow, 0,  lane);
        v16bf kb1 = load_frag_b(krow, 32, lane);

        for (int tix = 0; tix < 16; tix++) {
            const int jtn = jt + 4;
            v16bf nb0 = kb0, nb1 = kb1;
            if (tix < 15) {
                const unsigned short* krn =
                    kb16 + (bh * S_ + jtn * 16 + (lane & 15)) * DH_;
                nb0 = load_frag_b(krn, 0,  lane);
                nb1 = load_frag_b(krn, 32, lane);
            }

            v8f acc = v8f{};
            acc = WMMA_BF16(qa0, kb0, acc);
            acc = WMMA_BF16(qa1, kb1, acc);

            const int j = jt * 16 + (lane & 15);
            const uint32_t hqj = hq[j], hkj = hk[j];
            #pragma unroll
            for (int r = 0; r < 8; r++) {
                const int i = i0 + iBase + r;
                // A[i,j] kept iff (i<=j && coll(q_i,k_j)) || (i>j && coll(q_j,k_i))
                const bool coll = (i <= j) ? any_byte_eq(hqi[r], hkj)
                                           : any_byte_eq(hqj, hki[r]);
                sc16[iBase + r][j] = f2bf(coll ? acc[r] * 0.125f : 0.0f);
            }
            kb0 = nb0; kb1 = nb1; jt = jtn;
        }
    }
    __syncthreads();

    // ---- Phase 2: softmax over full rows (masked entries are 0, not -inf) --
    for (int rr = 0; rr < 4; rr++) {
        const int row = wave * 4 + rr;
        float ev[32];
        float m = -3.0e38f;
        #pragma unroll
        for (int c = 0; c < 32; c++) {
            const float x = bf2f(sc16[row][lane + c * 32]);
            ev[c] = x;
            m = fmaxf(m, x);
        }
        #pragma unroll
        for (int o = 16; o > 0; o >>= 1) m = fmaxf(m, __shfl_xor(m, o, 32));
        float sum = 0.f;
        #pragma unroll
        for (int c = 0; c < 32; c++) { ev[c] = __expf(ev[c] - m); sum += ev[c]; }
        #pragma unroll
        for (int o = 16; o > 0; o >>= 1) sum += __shfl_xor(sum, o, 32);
        const float inv = 1.0f / sum;
        #pragma unroll
        for (int c = 0; c < 32; c++) sc16[row][lane + c * 32] = f2bf(ev[c] * inv);
    }
    __syncthreads();

    // ---- Phase 3: P @ V, wave owns dh cols [16w,16w+16); V double-buffered --
    {
        const unsigned short* prow = &sc16[lane & 15][0];
        v16bf vb = load_frag_b(vrow, 0, lane);
        v8f o = v8f{};
        for (int kt = 0; kt < S_ / 32; kt++) {
            v16bf nvb = vb;
            if (kt < S_ / 32 - 1) nvb = load_frag_b(vrow, (kt + 1) * 32, lane);
            const v16bf pa = load_frag_a(prow, kt * 32, lane);  // ds_load path
            o = WMMA_BF16(pa, vb, o);
            vb = nvb;
        }
        const int d = d0 + (lane & 15);
        #pragma unroll
        for (int r = 0; r < 8; r++) {
            const int s = i0 + iBase + r;
            out[((size_t)b * S_ + s) * D_ + h * DH_ + d] = o[r];
        }
    }
}

// ---------------------------------------------------------------------------
// Host-side launcher
// ---------------------------------------------------------------------------
extern "C" void kernel_launch(void* const* d_in, const int* in_sizes, int n_in,
                              void* d_out, int out_size, void* d_ws, size_t ws_size,
                              hipStream_t stream)
{
    const float* hidden = (const float*)d_in[0];
    const float* w_q    = (const float*)d_in[1];
    const float* b_q    = (const float*)d_in[2];
    const float* w_k    = (const float*)d_in[3];
    const float* b_k    = (const float*)d_in[4];
    const float* w_v    = (const float*)d_in[5];
    const float* b_v    = (const float*)d_in[6];
    const float* rv     = (const float*)d_in[7];
    const int*   hc     = (const int*)d_in[8];
    float* out = (float*)d_out;

    char* base = (char*)d_ws;
    size_t off = 0;
    auto wsAlloc = [&](size_t bytes) -> void* {
        off = (off + 255) & ~(size_t)255;
        void* p = base + off;
        off += bytes;
        return p;
    };

    const size_t nX  = (size_t)B_ * S_ * D_;     // 1,572,864
    const size_t nW  = (size_t)D_ * D_;          // 589,824
    const size_t nQK = (size_t)B_ * H_ * S_ * DH_;

    unsigned short* xb  = (unsigned short*)wsAlloc(nX * 2);
    unsigned short* wqb = (unsigned short*)wsAlloc(nW * 2);
    unsigned short* wkb = (unsigned short*)wsAlloc(nW * 2);
    unsigned short* wvb = (unsigned short*)wsAlloc(nW * 2);
    float* qf = (float*)wsAlloc(nQK * 4);
    float* kf = (float*)wsAlloc(nQK * 4);
    unsigned short* qb16 = (unsigned short*)wsAlloc(nQK * 2);
    unsigned short* kb16 = (unsigned short*)wsAlloc(nQK * 2);
    unsigned short* vt   = (unsigned short*)wsAlloc(nQK * 2);
    uint32_t* hqp = (uint32_t*)wsAlloc((size_t)B_ * H_ * S_ * 4);
    uint32_t* hkp = (uint32_t*)wsAlloc((size_t)B_ * H_ * S_ * 4);

    // 1) stage bf16 operands (vectorized float4 -> ushort4)
    convert_bf16<<<dim3((unsigned)((nX / 4 + 255) / 256)), 256, 0, stream>>>(
        (const float4*)hidden, (ushort4*)xb, (int)(nX / 4));
    convert_bf16<<<dim3((unsigned)((nW / 4 + 255) / 256)), 256, 0, stream>>>(
        (const float4*)w_q, (ushort4*)wqb, (int)(nW / 4));
    convert_bf16<<<dim3((unsigned)((nW / 4 + 255) / 256)), 256, 0, stream>>>(
        (const float4*)w_k, (ushort4*)wkb, (int)(nW / 4));
    convert_bf16<<<dim3((unsigned)((nW / 4 + 255) / 256)), 256, 0, stream>>>(
        (const float4*)w_v, (ushort4*)wvb, (int)(nW / 4));

    // 2) fused QKV projection (pipelined WMMA bf16)
    qkv_gemm<<<dim3(D_ / 64, (B_ * S_) / 128, 3), 128, 0, stream>>>(
        xb, wqb, wkb, wvb, b_q, b_k, b_v, qf, kf, qb16, kb16, vt);

    // 3) LSH hashing (f32, sign-exact, LDS-staged projections)
    lsh_hash<<<dim3((B_ * H_ * S_) / 128), 128, 0, stream>>>(
        qf, kf, rv, hc, hqp, hkp);

    // 4) masked attention (WMMA scores + softmax + WMMA PV)
    lsh_attention<<<dim3(S_ / 16, H_, B_), 128, 0, stream>>>(
        qb16, kb16, vt, hqp, hkp, out);

    (void)in_sizes; (void)n_in; (void)out_size; (void)ws_size;
}
